// ExpertChoiceFF_36739150250479
// MI455X (gfx1250) — matrix operand order repaired and checked by reading, not compiled, not run on hardware
//
#include <hip/hip_runtime.h>

// ---------------------------------------------------------------------------
// ExpertChoiceFF for MI455X (gfx1250): gate path is dead code w.r.t. output.
// out = (relu(x_e @ W1_e)) @ W2_e  per expert, bf16 WMMA with f32 accumulate.
// ---------------------------------------------------------------------------

typedef __attribute__((ext_vector_type(16))) __bf16        v16bf;
typedef __attribute__((ext_vector_type(4)))  __bf16        bf16x4;
typedef __attribute__((ext_vector_type(8)))  float         v8f;
typedef __attribute__((ext_vector_type(4)))  float         f32x4;
typedef __attribute__((ext_vector_type(4)))  unsigned int  u32x4;

#define DMODEL 1024
#define ESIZE  2048
#define ROWS   256   // tokens per expert (batch*cutoff / n_experts)
#define NEXP   32

#define BM 128
#define BN 128
#define BK 32
#define LDSS 40      // padded LDS row stride (elements); 80B rows, 16B-aligned

union Frag { v16bf v; u32x4 q[2]; };

// Stage a 32x128 f32 tile of B into LDS as bf16, transposed to [N][K].
// Each thread owns a 4(K) x 4(N) micro-tile: 4 coalesced f32x4 loads,
// register transpose, 4 packed b64 LDS stores.
__device__ __forceinline__ void stage_b_tile(const float* __restrict__ B,
                                             int ldb, int k0, int n0, int KT,
                                             int tid, __bf16* sB) {
  const int nb = tid & 31;    // n block 0..31
  const int kb = tid >> 5;    // k block 0..7
  const int n  = nb << 2;
  const int k  = kb << 2;
  f32x4 vv[4];
  const float* gp = B + (size_t)(k0 + k) * ldb + n0 + n;
  #pragma unroll
  for (int r = 0; r < 4; ++r)
    vv[r] = *(const f32x4*)(gp + (size_t)r * ldb);
  if (k0 + BK < KT * BK) __builtin_prefetch(gp + (size_t)BK * ldb, 0, 3);
  #pragma unroll
  for (int c = 0; c < 4; ++c) {
    bf16x4 p = { (__bf16)vv[0][c], (__bf16)vv[1][c],
                 (__bf16)vv[2][c], (__bf16)vv[3][c] };
    *(bf16x4*)&sB[(n + c) * LDSS + k] = p;
  }
}

// ---------------------------------------------------------------------------
// GEMM1: H[e] = relu(X[e] (256x1024, f32) @ W1[e] (1024x2048, f32)) -> bf16
// grid = (ESIZE/BN, ROWS/BM, NEXP), block = 256
// ---------------------------------------------------------------------------
__global__ __launch_bounds__(256)
void moe_ffn1_kernel(const float* __restrict__ X, const float* __restrict__ W1,
                     __bf16* __restrict__ H) {
  __shared__ __bf16 sA[BM * LDSS];  // [row][k]
  __shared__ __bf16 sB[BN * LDSS];  // [col][k] (transposed)

  const int tid  = threadIdx.x;
  const int lane = tid & 31;
  const int wave = tid >> 5;
  const int wm   = wave & 3;    // 4 M-groups of 32 rows
  const int wn   = wave >> 2;   // 2 N-groups of 64 cols
  const int e    = blockIdx.z;
  const int m0   = blockIdx.y * BM;
  const int n0   = blockIdx.x * BN;

  const float* A = X  + (size_t)e * ROWS * DMODEL;
  const float* B = W1 + (size_t)e * DMODEL * ESIZE;
  __bf16* Hout   = H  + (size_t)e * ROWS * ESIZE;

  const int half = lane >> 4;   // 0: K 0-7,16-23; 1: K 8-15,24-31
  const int l16  = lane & 15;

  v8f acc[2][4] = {};

  const int KT = DMODEL / BK;
  for (int kt = 0; kt < KT; ++kt) {
    const int k0 = kt * BK;

    // ---- stage A tile: 128x32 f32 -> bf16 LDS (packed b64 stores) ----
    #pragma unroll
    for (int i = 0; i < 4; ++i) {
      int idx = tid + i * 256;          // 1024 float4 chunks
      int r   = idx >> 3;               // 0..127
      int c   = (idx & 7) << 2;         // 0,4,...,28
      f32x4 v = *(const f32x4*)(A + (size_t)(m0 + r) * DMODEL + k0 + c);
      bf16x4 p = { (__bf16)v.x, (__bf16)v.y, (__bf16)v.z, (__bf16)v.w };
      *(bf16x4*)&sA[r * LDSS + c] = p;
    }
    // ---- stage B tile: 32x128 f32 -> bf16 LDS, transposed to [N][K] ----
    stage_b_tile(B, ESIZE, k0, n0, KT, tid, sB);
    __syncthreads();

    // ---- fragments per ISA 16-bit layouts ----
    Frag a[2];
    #pragma unroll
    for (int i = 0; i < 2; ++i) {
      const __bf16* p = &sA[(wm * 32 + i * 16 + l16) * LDSS + half * 8];
      a[i].q[0] = *(const u32x4*)(p);        // K 0-7   / 8-15
      a[i].q[1] = *(const u32x4*)(p + 16);   // K 16-23 / 24-31
    }
    Frag b[4];
    #pragma unroll
    for (int j = 0; j < 4; ++j) {
      const __bf16* p = &sB[(wn * 64 + j * 16 + l16) * LDSS + half * 8];
      b[j].q[0] = *(const u32x4*)(p);        // K 0-7   / 8-15
      b[j].q[1] = *(const u32x4*)(p + 16);   // K 16-23 / 24-31
    }
    #pragma unroll
    for (int i = 0; i < 2; ++i)
      #pragma unroll
      for (int j = 0; j < 4; ++j)
        acc[i][j] = __builtin_amdgcn_wmma_f32_16x16x32_bf16(
            false, a[i].v, false, b[j].v, (short)0, acc[i][j], false, false);
    __syncthreads();
  }

  // ---- epilogue: relu + bf16 store (C/D layout: VGPR r -> M=r or r+8) ----
  #pragma unroll
  for (int i = 0; i < 2; ++i) {
    int rbase = m0 + wm * 32 + i * 16 + (half << 3);
    #pragma unroll
    for (int j = 0; j < 4; ++j) {
      int col = n0 + wn * 64 + j * 16 + l16;
      #pragma unroll
      for (int r = 0; r < 8; ++r) {
        float v = acc[i][j][r];
        v = v > 0.0f ? v : 0.0f;
        Hout[(size_t)(rbase + r) * ESIZE + col] = (__bf16)v;
      }
    }
  }
}

// ---------------------------------------------------------------------------
// GEMM2: OUT[e] = H[e] (256x2048, bf16) @ W2[e] (2048x1024, f32) -> f32
// grid = (DMODEL/BN, ROWS/BM, NEXP), block = 256
// ---------------------------------------------------------------------------
__global__ __launch_bounds__(256)
void moe_ffn2_kernel(const __bf16* __restrict__ Hin, const float* __restrict__ W2,
                     float* __restrict__ Out) {
  __shared__ __bf16 sA[BM * LDSS];
  __shared__ __bf16 sB[BN * LDSS];

  const int tid  = threadIdx.x;
  const int lane = tid & 31;
  const int wave = tid >> 5;
  const int wm   = wave & 3;
  const int wn   = wave >> 2;
  const int e    = blockIdx.z;
  const int m0   = blockIdx.y * BM;
  const int n0   = blockIdx.x * BN;

  const __bf16* A = Hin + (size_t)e * ROWS * ESIZE;
  const float*  B = W2  + (size_t)e * ESIZE * DMODEL;
  float* O        = Out + (size_t)e * ROWS * DMODEL;

  const int half = lane >> 4;
  const int l16  = lane & 15;

  v8f acc[2][4] = {};

  const int KT = ESIZE / BK;
  for (int kt = 0; kt < KT; ++kt) {
    const int k0 = kt * BK;

    // ---- stage A tile: 128x32 bf16, direct b128 copy ----
    #pragma unroll
    for (int i = 0; i < 2; ++i) {
      int idx = tid + i * 256;          // 512 chunks of 8 bf16
      int r   = idx >> 2;               // 0..127
      int c   = (idx & 3) << 3;         // 0,8,16,24
      u32x4 v = *(const u32x4*)(A + (size_t)(m0 + r) * ESIZE + k0 + c);
      *(u32x4*)&sA[r * LDSS + c] = v;
    }
    // ---- stage B tile: 32x128 f32 -> bf16 LDS, transposed to [N][K] ----
    stage_b_tile(B, DMODEL, k0, n0, KT, tid, sB);
    __syncthreads();

    Frag a[2];
    #pragma unroll
    for (int i = 0; i < 2; ++i) {
      const __bf16* p = &sA[(wm * 32 + i * 16 + l16) * LDSS + half * 8];
      a[i].q[0] = *(const u32x4*)(p);
      a[i].q[1] = *(const u32x4*)(p + 16);
    }
    Frag b[4];
    #pragma unroll
    for (int j = 0; j < 4; ++j) {
      const __bf16* p = &sB[(wn * 64 + j * 16 + l16) * LDSS + half * 8];
      b[j].q[0] = *(const u32x4*)(p);
      b[j].q[1] = *(const u32x4*)(p + 16);
    }
    #pragma unroll
    for (int i = 0; i < 2; ++i)
      #pragma unroll
      for (int j = 0; j < 4; ++j)
        acc[i][j] = __builtin_amdgcn_wmma_f32_16x16x32_bf16(
            false, a[i].v, false, b[j].v, (short)0, acc[i][j], false, false);
    __syncthreads();
  }

  // ---- epilogue: f32 nontemporal store (write-once, never re-read) ----
  #pragma unroll
  for (int i = 0; i < 2; ++i) {
    int rbase = m0 + wm * 32 + i * 16 + (half << 3);
    #pragma unroll
    for (int j = 0; j < 4; ++j) {
      int col = n0 + wn * 64 + j * 16 + l16;
      #pragma unroll
      for (int r = 0; r < 8; ++r)
        __builtin_nontemporal_store(acc[i][j][r],
                                    &O[(size_t)(rbase + r) * DMODEL + col]);
    }
  }
}

extern "C" void kernel_launch(void* const* d_in, const int* in_sizes, int n_in,
                              void* d_out, int out_size, void* d_ws, size_t ws_size,
                              hipStream_t stream) {
  const float* x  = (const float*)d_in[0];
  // d_in[1] = gate: dead code w.r.t. the reference's return value (unused)
  const float* w1 = (const float*)d_in[2];
  const float* w2 = (const float*)d_in[3];
  float* out      = (float*)d_out;
  __bf16* h       = (__bf16*)d_ws;   // 32 experts * 256 * 2048 * 2B = 32 MB

  dim3 blk(256);
  dim3 g1(ESIZE / BN, ROWS / BM, NEXP);   // (16, 2, 32)
  dim3 g2(DMODEL / BN, ROWS / BM, NEXP);  // (8, 2, 32)
  moe_ffn1_kernel<<<g1, blk, 0, stream>>>(x, w1, h);
  moe_ffn2_kernel<<<g2, blk, 0, stream>>>(h, w2, out);
}
